// CosineCentroidLoss_1846835938212
// MI455X (gfx1250) — compile-verified, matrix operand order checked by reading, not run on hardware
//
#include <hip/hip_runtime.h>
#include <hip/hip_bf16.h>
#include <cstddef>

typedef _Float16 v16h __attribute__((ext_vector_type(16)));
typedef float    v8f  __attribute__((ext_vector_type(8)));
typedef int      vi4  __attribute__((vector_size(16)));

#define C_CLS   1024
#define S_SMP   16
#define D_DIM   512
#define M_TOT   (C_CLS * S_SMP)           // 16384
#define KCHUNKS (D_DIM / 32)              // 16
#define NTILES  (C_CLS / 16)              // 64

// dword sizes of workspace regions
#define XNF_DW  ((size_t)C_CLS * 16 * 32 * 8)   // 4,194,304 dwords = 16 MB (A fragments, f16)
#define MF_DW   ((size_t)NTILES * 16 * 32 * 8)  //   262,144 dwords =  1 MB (B fragments, f16)

#if __has_builtin(__builtin_amdgcn_global_load_async_to_lds_b128)
#define HAS_ASYNC_LDS 1
#else
#define HAS_ASYNC_LDS 0
#endif

union Frag {
    v16h  v;
    uint4 q[2];
};

static __device__ __forceinline__ unsigned pack2h(float a, float b) {
    union { _Float16 h[2]; unsigned u; } x;
    x.h[0] = (_Float16)a;
    x.h[1] = (_Float16)b;
    return x.u;
}

// K-pair base index inside a 32-wide K chunk for (lane, vgpr) per the
// 16-bit A-matrix 16x32 layout (CDNA5 ISA 7.12.2). Lanes 0-15: v0..3 -> K0..7,
// v4..7 -> K16..23; lanes 16-31: v0..3 -> K8..15, v4..7 -> K24..31.
static __device__ __forceinline__ int kpair(int lane, int v) {
    int basek = (lane >= 16) ? 8 : 0;
    return ((v >> 2) * 16) + basek + ((v & 3) * 2);
}

static __device__ __forceinline__ void wait_async0() {
#if __has_builtin(__builtin_amdgcn_s_wait_asynccnt)
    __builtin_amdgcn_s_wait_asynccnt(0);
#else
    asm volatile("s_wait_asynccnt 0" ::: "memory");
#endif
}

#if HAS_ASYNC_LDS
// Stage one 16KB B tile (4096 dwords) global -> LDS via async direct-to-LDS
// copies (ASYNCcnt path). 256 threads x 4 x b128 each.
static __device__ __forceinline__ void stage_tile_async(const unsigned* __restrict__ src,
                                                        unsigned* dst, int tid) {
    #pragma unroll
    for (int t = 0; t < 4; ++t) {
        const int o = (tid + t * 256) * 4;
        __builtin_amdgcn_global_load_async_to_lds_b128(
            (__attribute__((address_space(1))) vi4*)(src + o),
            (__attribute__((address_space(3))) vi4*)(dst + o),
            0, 0);
    }
}
#endif

// ---------------------------------------------------------------------------
// Kernel 1: per-class prep. One block per class j.
// ---------------------------------------------------------------------------
__global__ __launch_bounds__(256) void prep_kernel(const float* __restrict__ x,
                                                   unsigned* __restrict__ xnf,
                                                   unsigned* __restrict__ mf,
                                                   float* __restrict__ simpos) {
    const int j    = blockIdx.x;
    const int tid  = threadIdx.x;
    const int lane = tid & 31;
    const int wave = tid >> 5;

    __shared__ __align__(16) float ssum[D_DIM];
    __shared__ float sinvnx[S_SMP];
    __shared__ float sinv_nsum;

    const float* xj = x + (size_t)j * (S_SMP * D_DIM);

    // column sums over the 16 samples
    for (int d = tid; d < D_DIM; d += 256) {
        float s = 0.f;
        #pragma unroll
        for (int i = 0; i < S_SMP; ++i) s += xj[i * D_DIM + d];
        ssum[d] = s;
    }
    __syncthreads();

    // ||sum||  (wave 0)
    if (wave == 0) {
        float p = 0.f;
        for (int d = lane; d < D_DIM; d += 32) { float s = ssum[d]; p += s * s; }
        #pragma unroll
        for (int off = 16; off >= 1; off >>= 1) p += __shfl_xor(p, off);
        if (lane == 0) sinv_nsum = 1.0f / fmaxf(sqrtf(p), 1e-12f);
    }

    // per-sample norms + exact leave-one-out cosine (wave w handles i=2w,2w+1)
    #pragma unroll
    for (int ii = 0; ii < 2; ++ii) {
        const int i = wave * 2 + ii;
        float a2 = 0.f, xs = 0.f, d2 = 0.f;
        for (int d = lane; d < D_DIM; d += 32) {
            float v = xj[i * D_DIM + d];
            float s = ssum[d];
            a2 += v * v;
            xs += v * s;
            float t = s - v;
            d2 += t * t;
        }
        #pragma unroll
        for (int off = 16; off >= 1; off >>= 1) {
            a2 += __shfl_xor(a2, off);
            xs += __shfl_xor(xs, off);
            d2 += __shfl_xor(d2, off);
        }
        if (lane == 0) {
            float nx  = sqrtf(a2);
            float num = (xs - a2) * (1.0f / 15.0f);
            float den = fmaxf(nx * sqrtf(d2) * (1.0f / 15.0f), 1e-8f);
            simpos[j * S_SMP + i] = num / den;
            sinvnx[i] = 1.0f / fmaxf(nx, 1e-12f);
        }
    }
    __syncthreads();

    // normalized x -> A fragments (f16, fragment-order so kernel 2 loads 32B/lane)
    #pragma unroll
    for (int it = 0; it < 2; ++it) {
        const int flat = it * 256 + tid;        // 0..511 = (kc, L)
        const int kc   = flat >> 5;
        const int L    = flat & 31;
        const int i    = L & 15;
        const float inv = sinvnx[i];
        unsigned out[8];
        #pragma unroll
        for (int v = 0; v < 8; ++v) {
            const int k0 = kc * 32 + kpair(L, v);
            out[v] = pack2h(xj[i * D_DIM + k0] * inv, xj[i * D_DIM + k0 + 1] * inv);
        }
        unsigned* dst = xnf + (((size_t)(j * 16 + kc) * 32) + L) * 8;
        ((uint4*)dst)[0] = make_uint4(out[0], out[1], out[2], out[3]);
        ((uint4*)dst)[1] = make_uint4(out[4], out[5], out[6], out[7]);
    }

    // normalized centroid -> B fragment entries (column j&15 of tile j>>4)
    {
        const int kc = tid >> 4;        // 0..15
        const int lh = (tid >> 3) & 1;  // lane half
        const int v  = tid & 7;
        const int L  = lh * 16 + (j & 15);
        const int k0 = kc * 32 + kpair(L, v);
        const float inv = sinv_nsum;
        const int nt = j >> 4;
        mf[(((size_t)(nt * 16 + kc) * 32) + L) * 8 + v] =
            pack2h(ssum[k0] * inv, ssum[k0 + 1] * inv);
    }
}

// ---------------------------------------------------------------------------
// Per-tile compute: 16 pipelined ds_load_b128x2 + v_wmma_f32_16x16x32_f16,
// exact-diagonal substitution, online logsumexp update.
// ---------------------------------------------------------------------------
static __device__ __forceinline__ void tile_compute(const v16h* __restrict__ a,
                                                    const unsigned* cur,
                                                    int lane, int nt, int j,
                                                    float wp, float bb,
                                                    const float* __restrict__ sp,
                                                    float* __restrict__ mx,
                                                    float* __restrict__ sm) {
    v8f acc = {0.f, 0.f, 0.f, 0.f, 0.f, 0.f, 0.f, 0.f};
    Frag fcur, fnxt;
    {
        const uint4* bp = (const uint4*)(cur + lane * 8);
        fcur.q[0] = bp[0];
        fcur.q[1] = bp[1];
    }
    #pragma unroll
    for (int kc = 0; kc < KCHUNKS; ++kc) {
        if (kc + 1 < KCHUNKS) {
            const uint4* bp = (const uint4*)(cur + (kc + 1) * 256 + lane * 8);
            fnxt.q[0] = bp[0];
            fnxt.q[1] = bp[1];
        }
        acc = __builtin_amdgcn_wmma_f32_16x16x32_f16(
            false, a[kc], false, fcur.v, (short)0, acc, false, false);
        fcur = fnxt;
    }

    const int  kcol   = nt * 16 + (lane & 15);
    const bool isdiag = (kcol == j);
    #pragma unroll
    for (int r = 0; r < 8; ++r) {
        float sim = isdiag ? sp[r] : acc[r];
        float lg  = wp * sim + bb;
        float mn  = fmaxf(mx[r], lg);
        sm[r] = sm[r] * __expf(mx[r] - mn) + __expf(lg - mn);
        mx[r] = mn;
    }
}

// ---------------------------------------------------------------------------
// Kernel 2: WMMA GEMM (16384x1024x512, f16 in / f32 acc) fused with online
// logsumexp. 8 waves/block; wave w owns class j = blockIdx*8+w. B tiles are
// double-buffered in LDS and staged with async direct-to-LDS copies that
// overlap the WMMA work on the current tile.
// ---------------------------------------------------------------------------
__global__ __launch_bounds__(256) void gemm_lse_kernel(const unsigned* __restrict__ xnf,
                                                       const unsigned* __restrict__ mf,
                                                       const float* __restrict__ simpos,
                                                       const float* __restrict__ wptr,
                                                       const float* __restrict__ bptr,
                                                       float* __restrict__ partials) {
    __shared__ __align__(16) unsigned btile[2 * 16 * 32 * 8];   // 2 x 16 KB
    __shared__ float wpart[8];

    const int tid  = threadIdx.x;
    const int wave = tid >> 5;
    const int lane = tid & 31;
    const int j    = blockIdx.x * 8 + wave;

    const float w  = wptr[0];
    const float bb = bptr[0];
    const float wp = fmaxf(w, 0.0f) + log1pf(__expf(-fabsf(w)));   // softplus

    // Load full A strip for this class: 16 K-chunk fragments (128 VGPRs)
    v16h a[KCHUNKS];
    {
        const unsigned* abase = xnf + (size_t)j * 4096 + lane * 8;
        #pragma unroll
        for (int kc = 0; kc < KCHUNKS; ++kc) {
            Frag f;
            const uint4* p = (const uint4*)(abase + kc * 256);
            f.q[0] = p[0];
            f.q[1] = p[1];
            a[kc] = f.v;
        }
    }

    // Exact diagonal logits + substitution values (same across lanes in a half)
    float sp[8], diag[8];
    #pragma unroll
    for (int r = 0; r < 8; ++r) {
        const int irow = r + 8 * (lane >> 4);
        sp[r]   = simpos[j * S_SMP + irow];
        diag[r] = wp * sp[r] + bb;
    }

    // online logsumexp state (per lane: its 1/16 share of each row's columns)
    float mx[8], sm[8];
    #pragma unroll
    for (int r = 0; r < 8; ++r) { mx[r] = -3.0e38f; sm[r] = 0.0f; }

#if HAS_ASYNC_LDS
    // prologue: stage tile 0
    stage_tile_async(mf, btile, tid);
    wait_async0();
    __syncthreads();
    for (int nt = 0; nt < NTILES; ++nt) {
        const unsigned* cur = btile + (nt & 1) * 4096;
        if (nt + 1 < NTILES)
            stage_tile_async(mf + (size_t)(nt + 1) * 4096,
                             btile + ((nt + 1) & 1) * 4096, tid);
        tile_compute(a, cur, lane, nt, j, wp, bb, sp, mx, sm);
        wait_async0();
        __syncthreads();
    }
#else
    for (int nt = 0; nt < NTILES; ++nt) {
        __syncthreads();
        {
            const uint4* src = (const uint4*)(mf + (size_t)nt * 4096);
            uint4* dst = (uint4*)btile;
            #pragma unroll
            for (int t = 0; t < 4; ++t) dst[tid + t * 256] = src[tid + t * 256];
        }
        __syncthreads();
        tile_compute(a, btile, lane, nt, j, wp, bb, sp, mx, sm);
    }
#endif

    // combine partial (max,sum) across the 16 lanes of each half
    #pragma unroll
    for (int off = 8; off >= 1; off >>= 1) {
        #pragma unroll
        for (int r = 0; r < 8; ++r) {
            float om = __shfl_xor(mx[r], off);
            float os = __shfl_xor(sm[r], off);
            float mn = fmaxf(mx[r], om);
            sm[r] = sm[r] * __expf(mx[r] - mn) + os * __expf(om - mn);
            mx[r] = mn;
        }
    }

    // per-row loss = logsumexp(logits) - diag_logit ; sum rows of this class
    float lsum = 0.0f;
    if ((lane & 15) == 0) {
        #pragma unroll
        for (int r = 0; r < 8; ++r) lsum += (mx[r] + logf(sm[r])) - diag[r];
    }
    #pragma unroll
    for (int off = 16; off >= 1; off >>= 1) lsum += __shfl_xor(lsum, off);
    if (lane == 0) wpart[wave] = lsum;
    __syncthreads();
    if (tid == 0) {
        float s = 0.0f;
        #pragma unroll
        for (int t = 0; t < 8; ++t) s += wpart[t];
        partials[blockIdx.x] = s;
    }
}

// ---------------------------------------------------------------------------
// Kernel 3: deterministic final reduction (128 partials -> mean loss)
// ---------------------------------------------------------------------------
__global__ __launch_bounds__(128) void finalize_kernel(const float* __restrict__ partials,
                                                       float* __restrict__ out) {
    __shared__ float s4[4];
    const int tid  = threadIdx.x;
    const int lane = tid & 31;
    const int wave = tid >> 5;
    float v = partials[tid];
    #pragma unroll
    for (int off = 16; off >= 1; off >>= 1) v += __shfl_xor(v, off);
    if (lane == 0) s4[wave] = v;
    __syncthreads();
    if (tid == 0) out[0] = (s4[0] + s4[1] + s4[2] + s4[3]) * (1.0f / (float)M_TOT);
}

extern "C" void kernel_launch(void* const* d_in, const int* in_sizes, int n_in,
                              void* d_out, int out_size, void* d_ws, size_t ws_size,
                              hipStream_t stream) {
    const float* x = (const float*)d_in[0];
    const float* w = (const float*)d_in[1];
    const float* b = (const float*)d_in[2];

    unsigned* xnf      = (unsigned*)d_ws;
    unsigned* mf       = xnf + XNF_DW;
    float*    simpos   = (float*)(mf + MF_DW);
    float*    partials = simpos + M_TOT;

    prep_kernel<<<C_CLS, 256, 0, stream>>>(x, xnf, mf, simpos);
    gemm_lse_kernel<<<M_TOT / (16 * 8), 256, 0, stream>>>(xnf, mf, simpos, w, b, partials);
    finalize_kernel<<<1, 128, 0, stream>>>(partials, (float*)d_out);
}
